// NeuralODEEncoderRNN_44805098832134
// MI455X (gfx1250) — matrix operand order repaired and checked by reading, not compiled, not run on hardware
//
#include <hip/hip_runtime.h>
#include <math.h>

typedef __bf16 bf16;
typedef __attribute__((ext_vector_type(16))) __bf16 v16bf;
typedef __attribute__((ext_vector_type(8)))  __bf16 v8bf;
typedef __attribute__((ext_vector_type(4)))  __bf16 v4bf;
typedef __attribute__((ext_vector_type(8)))  float  v8f;
typedef __attribute__((ext_vector_type(4)))  float  v4f;

#define IN_DIM  128
#define HID     256
#define LAT     64
#define NSTEP   16
#define BM      16      // batch rows per workgroup
#define LDSTR   264     // bf16 elements per LDS row (256 + 8 pad, breaks bank aliasing)
#define THREADS 256     // 8 wave32

// Dormand-Prince 5(4) tableau
constexpr float DP_A[5][5] = {
  {1.f/5.f, 0.f, 0.f, 0.f, 0.f},
  {3.f/40.f, 9.f/40.f, 0.f, 0.f, 0.f},
  {44.f/45.f, -56.f/15.f, 32.f/9.f, 0.f, 0.f},
  {19372.f/6561.f, -25360.f/2187.f, 64448.f/6561.f, -212.f/729.f, 0.f},
  {9017.f/3168.f, -355.f/33.f, 46732.f/5247.f, 49.f/176.f, -5103.f/18656.f},
};
constexpr float DP_B[6] = {35.f/384.f, 0.f, 500.f/1113.f, 125.f/192.f, -2187.f/6784.f, 11.f/84.f};
constexpr float DP_C[6] = {0.f, 1.f/5.f, 3.f/10.f, 4.f/5.f, 8.f/9.f, 1.f};

// CDNA5 has a hardware TANH transcendental (TRANS32, co-executes with WMMA).
static __device__ __forceinline__ float fast_tanh(float x) {
#if __has_builtin(__builtin_amdgcn_tanhf)
  return __builtin_amdgcn_tanhf(x);
#else
  return tanhf(x);
#endif
}

static __device__ __forceinline__ v8f wmma_bf16(v16bf a, v16bf b, v8f c) {
  // D = A(16x32 bf16) * B(32x16 bf16) + C(16x16 f32)
  return __builtin_amdgcn_wmma_f32_16x16x32_bf16(false, a, false, b, (short)0, c, false, false);
}

// A fragment (16x32, M = lane&15): lanes 0-15 take K = kb+[0..15],
// lanes 16-31 take K = kb+[16..31]  (matches B fragment k order)
static __device__ __forceinline__ v16bf ld_a_lds(const bf16* s, int row, int kb, int half) {
  const bf16* p = s + row * LDSTR + kb + half * 16;
  v8bf lo = *(const v8bf*)p;         // ds_load_b128
  v8bf hi = *(const v8bf*)(p + 8);   // ds_load_b128
  return __builtin_shufflevector(lo, hi, 0,1,2,3,4,5,6,7,8,9,10,11,12,13,14,15);
}
// B fragment (32x16, N = lane&15): lanes 0-15 hold K = kb+[0..15], lanes 16-31 K = kb+[16..31]
static __device__ __forceinline__ v16bf ld_b_lds(const bf16* s, int row, int kb, int half) {
  const bf16* p = s + row * LDSTR + kb + half * 16;
  v8bf lo = *(const v8bf*)p;
  v8bf hi = *(const v8bf*)(p + 8);
  return __builtin_shufflevector(lo, hi, 0,1,2,3,4,5,6,7,8,9,10,11,12,13,14,15);
}
// B fragment straight from global f32 (small, once-used weights), converted on the fly
static __device__ __forceinline__ v16bf ld_b_glb(const float* W, int ld, int row, int kb, int half) {
  const float* p = W + (size_t)row * ld + kb + half * 16;
  v16bf r;
  #pragma unroll
  for (int i = 0; i < 16; ++i) r[i] = (bf16)p[i];
  return r;
}
// C-fragment (f32, M = vgpr index + 8*lanehalf, N = lane&15) -> LDS bf16 [M][K] tile
static __device__ __forceinline__ void st_c_lds(bf16* s, int n, int mrow, v8f c) {
  #pragma unroll
  for (int v = 0; v < 8; ++v) s[(mrow + v) * LDSTR + n] = (bf16)c[v];
}

__global__ __launch_bounds__(THREADS, 1)
void node_fused(const float* __restrict__ x, const float* __restrict__ tp, int ntp,
                const float* __restrict__ W_in, const float* __restrict__ b_in,
                const float* __restrict__ W_ih, const float* __restrict__ b_ih,
                const float* __restrict__ b_hh, const float* __restrict__ W_out,
                const float* __restrict__ b_out, const float* __restrict__ W_lat,
                const float* __restrict__ b_lat, float* __restrict__ out) {
  // gfx1250: 320 KB LDS per WGP — stage both recurrent weight matrices on-chip.
  __shared__ __align__(16) bf16 sWih [HID * LDSTR];  // 135168 B, [N][K] bf16
  __shared__ __align__(16) bf16 sWout[HID * LDSTR];  // 135168 B, [N][K] bf16
  __shared__ __align__(16) bf16 sA   [BM  * LDSTR];  // stage-input tile (A of GEMM1)
  __shared__ __align__(16) bf16 sR   [BM  * LDSTR];  // tanh output tile (A of GEMM2)

  const int tid  = threadIdx.x;
  const int wv   = tid >> 5;
  const int lane = tid & 31;
  const int half = lane >> 4;
  const int l15  = lane & 15;
  const int m0   = blockIdx.x * BM;

  // ---- stage weights into LDS (f32 -> bf16), one row per thread ----
  {
    const float* src = W_ih + (size_t)tid * (HID + 1);  // drop t-column (folded into bias)
    bf16* dst = sWih + tid * LDSTR;
    #pragma unroll 8
    for (int k = 0; k < HID; ++k) dst[k] = (bf16)src[k];
  }
  {
    const float* src = W_out + (size_t)tid * HID;
    bf16* dst = sWout + tid * LDSTR;
    #pragma unroll 4
    for (int kc = 0; kc < HID; kc += 4) {
      v4f v = *(const v4f*)(src + kc);
      v4bf b; b[0]=(bf16)v[0]; b[1]=(bf16)v[1]; b[2]=(bf16)v[2]; b[3]=(bf16)v[3];
      *(v4bf*)(dst + kc) = b;
    }
  }
  // ---- stage x tile into sA (bf16) ----
  {
    const int mm = tid >> 4;          // 0..15
    const int kk = (tid & 15) * 8;    // 0..120
    const float* xs = x + (size_t)(m0 + mm) * IN_DIM + kk;
    v4f lo = *(const v4f*)xs;
    v4f hi = *(const v4f*)(xs + 4);
    v8bf b;
    b[0]=(bf16)lo[0]; b[1]=(bf16)lo[1]; b[2]=(bf16)lo[2]; b[3]=(bf16)lo[3];
    b[4]=(bf16)hi[0]; b[5]=(bf16)hi[1]; b[6]=(bf16)hi[2]; b[7]=(bf16)hi[3];
    *(v8bf*)(sA + mm * LDSTR + kk) = b;
  }
  __syncthreads();

  // wave w owns output columns [32w, 32w+32): two 16x16 fragments
  const int nA = wv * 32 + l15;
  const int nB = nA + 16;
  const int mrow = half * 8;

  // ---- input projection: h = x @ W_in^T + b_in ----
  v8f h0 = {}, h1 = {};
  #pragma unroll
  for (int kb = 0; kb < IN_DIM; kb += 32) {
    v16bf a = ld_a_lds(sA, l15, kb, half);
    h0 = wmma_bf16(a, ld_b_glb(W_in, IN_DIM, nA, kb, half), h0);
    h1 = wmma_bf16(a, ld_b_glb(W_in, IN_DIM, nB, kb, half), h1);
  }
  {
    const float bi0 = b_in[nA], bi1 = b_in[nB];
    #pragma unroll
    for (int v = 0; v < 8; ++v) { h0[v] += bi0; h1[v] += bi1; }
  }

  // per-column constants: bias core and folded t-column of W_ih
  const float bihh0 = b_ih[nA] + b_hh[nA];
  const float bihh1 = b_ih[nB] + b_hh[nB];
  const float wl0 = W_ih[(size_t)nA * (HID + 1) + HID];
  const float wl1 = W_ih[(size_t)nB * (HID + 1) + HID];
  const float bo0 = b_out[nA];
  const float bo1 = b_out[nB];

  const float tstart = tp[0];
  const float dtv = (tp[ntp - 1] - tstart) / (float)NSTEP;

  __syncthreads();  // all waves done reading x tile from sA

  // ---- DOPRI5 fixed-step loop; h and k1..k6 live in registers ----
  v8f kf[6][2];
  #pragma unroll 1
  for (int step = 0; step < NSTEP; ++step) {
    const float t = tstart + dtv * (float)step;
    #pragma unroll
    for (int s = 0; s < 6; ++s) {
      // stage input: g = h + dt * sum_j A[s-1][j] * k_j
      v8f g0 = h0, g1 = h1;
      #pragma unroll
      for (int j = 0; j < 5; ++j) {
        if (s > 0 && j < s) {
          const float w = dtv * DP_A[s - 1][j];
          #pragma unroll
          for (int v = 0; v < 8; ++v) { g0[v] += w * kf[j][0][v]; g1[v] += w * kf[j][1][v]; }
        }
      }
      st_c_lds(sA, nA, mrow, g0);
      st_c_lds(sA, nB, mrow, g1);
      __syncthreads();

      // GEMM1: r = tanh(g @ W_ih[:, :256]^T + (b_ih + b_hh + t_s * W_ih[:,256]))
      v8f r0 = {}, r1 = {};
      #pragma unroll
      for (int kb = 0; kb < HID; kb += 32) {
        v16bf a = ld_a_lds(sA, l15, kb, half);
        r0 = wmma_bf16(a, ld_b_lds(sWih, nA, kb, half), r0);
        r1 = wmma_bf16(a, ld_b_lds(sWih, nB, kb, half), r1);
      }
      const float ts  = t + DP_C[s] * dtv;
      const float bs0 = bihh0 + ts * wl0;
      const float bs1 = bihh1 + ts * wl1;
      #pragma unroll
      for (int v = 0; v < 8; ++v) {
        r0[v] = fast_tanh(r0[v] + bs0);   // v_tanh_f32 (TRANS32, co-executes w/ WMMA)
        r1[v] = fast_tanh(r1[v] + bs1);
      }
      st_c_lds(sR, nA, mrow, r0);
      st_c_lds(sR, nB, mrow, r1);
      __syncthreads();

      // GEMM2: k_s = r @ W_out^T + b_out  (accumulate straight into k fragments)
      v8f k0, k1;
      #pragma unroll
      for (int v = 0; v < 8; ++v) { k0[v] = bo0; k1[v] = bo1; }
      #pragma unroll
      for (int kb = 0; kb < HID; kb += 32) {
        v16bf a = ld_a_lds(sR, l15, kb, half);
        k0 = wmma_bf16(a, ld_b_lds(sWout, nA, kb, half), k0);
        k1 = wmma_bf16(a, ld_b_lds(sWout, nB, kb, half), k1);
      }
      kf[s][0] = k0;
      kf[s][1] = k1;
    }
    // h += dt * (b1 k1 + b3 k3 + b4 k4 + b5 k5 + b6 k6)   (b2 == 0)
    #pragma unroll
    for (int v = 0; v < 8; ++v) {
      h0[v] += dtv * (DP_B[0]*kf[0][0][v] + DP_B[2]*kf[2][0][v] + DP_B[3]*kf[3][0][v]
                    + DP_B[4]*kf[4][0][v] + DP_B[5]*kf[5][0][v]);
      h1[v] += dtv * (DP_B[0]*kf[0][1][v] + DP_B[2]*kf[2][1][v] + DP_B[3]*kf[3][1][v]
                    + DP_B[4]*kf[4][1][v] + DP_B[5]*kf[5][1][v]);
    }
  }

  // ---- latent projection: out = h @ W_lat^T + b_lat  (N = 64 -> waves 0..3) ----
  st_c_lds(sA, nA, mrow, h0);
  st_c_lds(sA, nB, mrow, h1);
  __syncthreads();
  if (wv < 4) {
    const int nl = wv * 16 + l15;
    v8f acc = {};
    #pragma unroll
    for (int kb = 0; kb < HID; kb += 32) {
      v16bf a = ld_a_lds(sA, l15, kb, half);
      acc = wmma_bf16(a, ld_b_glb(W_lat, HID, nl, kb, half), acc);
    }
    const float bl = b_lat[nl];
    #pragma unroll
    for (int v = 0; v < 8; ++v)
      out[(size_t)(m0 + mrow + v) * LAT + nl] = acc[v] + bl;
  }
}

extern "C" void kernel_launch(void* const* d_in, const int* in_sizes, int n_in,
                              void* d_out, int out_size, void* d_ws, size_t ws_size,
                              hipStream_t stream) {
  const float* x     = (const float*)d_in[0];
  const float* tp    = (const float*)d_in[1];
  const float* W_in  = (const float*)d_in[2];
  const float* b_in  = (const float*)d_in[3];
  const float* W_ih  = (const float*)d_in[4];
  const float* b_ih  = (const float*)d_in[5];
  const float* b_hh  = (const float*)d_in[6];
  const float* W_out = (const float*)d_in[7];
  const float* b_out = (const float*)d_in[8];
  const float* W_lat = (const float*)d_in[9];
  const float* b_lat = (const float*)d_in[10];
  float* out = (float*)d_out;

  const int rows = in_sizes[0] / IN_DIM;   // 16384
  const int ntp  = in_sizes[1];            // 8
  dim3 grid(rows / BM), block(THREADS);
  node_fused<<<grid, block, 0, stream>>>(x, tp, ntp, W_in, b_in, W_ih, b_ih, b_hh,
                                         W_out, b_out, W_lat, b_lat, out);
}
